// LoRACrossAttnProcessor_84293028151908
// MI455X (gfx1250) — compile-verified, matrix operand order checked by reading, not run on hardware
//
#include <hip/hip_runtime.h>
#include <cstdint>
#include <cstddef>

// ---------------------------------------------------------------------------
// Types for CDNA5 WMMA (gfx1250, wave32)
// ---------------------------------------------------------------------------
typedef __attribute__((ext_vector_type(16))) __bf16 bf16x16;
typedef __attribute__((ext_vector_type(8)))  __bf16 bf16x8;
typedef __attribute__((ext_vector_type(8)))  float  f32x8;

#define WMMA_BF16(a, b, c) \
  __builtin_amdgcn_wmma_f32_16x16x32_bf16(false, (a), false, (b), (short)0, (c), false, false)

__device__ __forceinline__ unsigned short f2bf(float f) {
  union { float f; unsigned u; } x; x.f = f;
  unsigned r = x.u + 0x7FFFu + ((x.u >> 16) & 1u);  // round-to-nearest-even
  return (unsigned short)(r >> 16);
}

// Load one 16x32 bf16 A/B^T fragment for v_wmma_f32_16x16x32_bf16.
// Lanes 0-15 hold K=0..7 / 16..23; lanes 16-31 hold K=8..15 / 24..31.
__device__ __forceinline__ bf16x16 ld_frag(const unsigned short* p) {
  bf16x8 lo = *reinterpret_cast<const bf16x8*>(p);
  bf16x8 hi = *reinterpret_cast<const bf16x8*>(p + 16);
  return __builtin_shufflevector(lo, hi, 0,1,2,3,4,5,6,7,8,9,10,11,12,13,14,15);
}

__device__ __forceinline__ f32x8 fzero8() {
  f32x8 z; for (int i = 0; i < 8; ++i) z[i] = 0.f; return z;
}

// Raw LDS (addrspace(3)) byte offset of a __shared__ pointer.
__device__ __forceinline__ unsigned lds_off(const void* p) {
  return (unsigned)(unsigned long long)
      (__attribute__((address_space(3))) const void*)p;
}

// CDNA5 async memory->LDS copy (no VGPR staging, tracked by ASYNCcnt).
__device__ __forceinline__ void async_load_b128(unsigned lds, const void* gptr) {
  asm volatile("global_load_async_to_lds_b128 %0, %1, off"
               :: "v"(lds), "v"(gptr) : "memory");
}
// Wait until ASYNCcnt <= 0 / <= 6 (6 = one 6-op tile still in flight).
__device__ __forceinline__ void wait_async0() {
  asm volatile("s_wait_asynccnt 0x0" ::: "memory");
}
__device__ __forceinline__ void wait_async6() {
  asm volatile("s_wait_asynccnt 0x6" ::: "memory");
}

// ---------------------------------------------------------------------------
// Problem constants (reference shapes)
// ---------------------------------------------------------------------------
#define BB   16
#define SS   1024
#define CC   1280
#define SENC 77
#define CENC 1024
#define HH   8
#define DH   160      // head dim
#define SKVP 96       // Senc padded to multiple of 32 for P@V

// ---------------------------------------------------------------------------
// Prep kernels
// ---------------------------------------------------------------------------
__global__ void fold_weight(const float* __restrict__ W,
                            const float* __restrict__ up,
                            const float* __restrict__ down,
                            unsigned short* __restrict__ out,
                            int N, int K) {
  int idx = blockIdx.x * blockDim.x + threadIdx.x;
  if (idx >= N * K) return;
  int n = idx / K, k = idx - n * K;
  float a = W[idx];
  #pragma unroll
  for (int r = 0; r < 4; ++r) a += up[n * 4 + r] * down[r * K + k];
  out[idx] = f2bf(a);
}

__global__ void f32_to_bf16(const float* __restrict__ in,
                            unsigned short* __restrict__ out, int n) {
  int i = blockIdx.x * blockDim.x + threadIdx.x;
  if (i < n) out[i] = f2bf(in[i]);
}

// vT[b][h][d][n] = (n < 77) ? v[(b*77+n)*1280 + h*160 + d] : 0
__global__ void build_vT(const unsigned short* __restrict__ vbuf,
                         unsigned short* __restrict__ vT) {
  int i = blockIdx.x * blockDim.x + threadIdx.x;
  const int total = BB * HH * DH * SKVP;
  if (i >= total) return;
  int n = i % SKVP; int t = i / SKVP;
  int d = t % DH;   t /= DH;
  int h = t % HH;   int b = t / HH;
  unsigned short v = 0;
  if (n < SENC) v = vbuf[((size_t)(b * SENC + n)) * CC + h * DH + d];
  vT[i] = v;
}

// ---------------------------------------------------------------------------
// Tiled bf16 GEMM:  Out[M,N] = A[M,K] @ W[N,K]^T   (+bias, fp32 out optional)
// Block 256 thr = 8 waves; block tile 128(M) x 256(N), K-step 32.
// Waves 2(M) x 4(N); wave tile 64x64 = 4x4 fragments (16 WMMA / K-step).
// Global->LDS async copies, TRIPLE-buffered: tile k+2 issued while computing
// tile k; partial wait s_wait_asynccnt<=6 keeps one tile in flight across the
// barrier, so prefetch latency is covered by ~2 full WMMA phases.
// ---------------------------------------------------------------------------
#define BM 128
#define BN 256
#define BK 32

__global__ __launch_bounds__(256)
void gemm_bf16(const unsigned short* __restrict__ A,
               const unsigned short* __restrict__ W,
               void* __restrict__ Out,
               const float* __restrict__ bias,
               int M, int N, int K, int outF32) {
  __shared__ alignas(16) unsigned short As[3][BM * BK];   // 3 x 8 KB
  __shared__ alignas(16) unsigned short Bs[3][BN * BK];   // 3 x 16 KB

  const int tid  = threadIdx.x;
  const int lane = tid & 31;
  const int wave = tid >> 5;
  const int wm   = wave >> 2;   // 0..1
  const int wn   = wave & 3;    // 0..3
  const int half = lane >> 4;   // 0..1
  const int l16  = lane & 15;

  const int bm0 = blockIdx.x * BM;
  const int bn0 = blockIdx.y * BN;

  f32x8 acc[4][4];
  #pragma unroll
  for (int i = 0; i < 4; ++i)
    #pragma unroll
    for (int j = 0; j < 4; ++j) acc[i][j] = fzero8();

  // Per-thread async-copy assignments (16B chunks).
  // A tile: 128x32 bf16 = 512 chunks -> 2/thread.  B tile: 1024 -> 4/thread.
  const unsigned short* aSrc[2]; unsigned aDst[2];
  #pragma unroll
  for (int i = 0; i < 2; ++i) {
    int c = tid + i * 256, r = c >> 2, col = (c & 3) * 8;
    int sr = bm0 + r; if (sr >= M) sr = M - 1;
    aSrc[i] = A + (size_t)sr * K + col;
    aDst[i] = (unsigned)((r * BK + col) * 2);
  }
  const unsigned short* bSrc[4]; unsigned bDst[4];
  #pragma unroll
  for (int i = 0; i < 4; ++i) {
    int c = tid + i * 256, r = c >> 2, col = (c & 3) * 8;
    bSrc[i] = W + (size_t)(bn0 + r) * K + col;
    bDst[i] = (unsigned)((r * BK + col) * 2);
  }
  const unsigned asB[3] = { lds_off(&As[0][0]), lds_off(&As[1][0]), lds_off(&As[2][0]) };
  const unsigned bsB[3] = { lds_off(&Bs[0][0]), lds_off(&Bs[1][0]), lds_off(&Bs[2][0]) };

  auto issue_tile = [&](int kElem, int buf) {   // 6 async ops per wave
    #pragma unroll
    for (int i = 0; i < 2; ++i) async_load_b128(asB[buf] + aDst[i], aSrc[i] + kElem);
    #pragma unroll
    for (int i = 0; i < 4; ++i) async_load_b128(bsB[buf] + bDst[i], bSrc[i] + kElem);
  };

  const int nk = K / BK;
  issue_tile(0, 0);
  if (nk > 1) { issue_tile(BK, 1); wait_async6(); }
  else        { wait_async0(); }
  __syncthreads();

  int cb = 0;  // compute buffer = kk % 3
  for (int kk = 0; kk < nk; ++kk) {
    const bool more  = (kk + 1 < nk);
    const bool issue = (kk + 2 < nk);
    if (issue) {
      int lb = cb + 2; if (lb >= 3) lb -= 3;
      issue_tile((kk + 2) * BK, lb);
    }

    // Hoist all fragment reads, then a WMMA burst.
    bf16x16 af[4], bf[4];
    #pragma unroll
    for (int am = 0; am < 4; ++am)
      af[am] = ld_frag(&As[cb][(wm * 64 + am * 16 + l16) * BK + half * 8]);
    #pragma unroll
    for (int bn = 0; bn < 4; ++bn)
      bf[bn] = ld_frag(&Bs[cb][(wn * 64 + bn * 16 + l16) * BK + half * 8]);
    #pragma unroll
    for (int am = 0; am < 4; ++am)
      #pragma unroll
      for (int bn = 0; bn < 4; ++bn)
        acc[am][bn] = WMMA_BF16(af[am], bf[bn], acc[am][bn]);

    if (more) {
      if (issue) wait_async6();   // oldest tile (k+1) resident; k+2 may fly
      else       wait_async0();   // tail: nothing newer in flight
      __syncthreads();
    }
    if (++cb == 3) cb = 0;
  }

  // D layout: VGPR j, lanes 0-15 -> row j, lanes 16-31 -> row j+8; col = l16.
  #pragma unroll
  for (int am = 0; am < 4; ++am) {
    #pragma unroll
    for (int bn = 0; bn < 4; ++bn) {
      const int col = bn0 + wn * 64 + bn * 16 + l16;
      #pragma unroll
      for (int j = 0; j < 8; ++j) {
        const int row = bm0 + wm * 64 + am * 16 + j + half * 8;
        if (row < M) {
          const float v = acc[am][bn][j];
          if (outF32)
            reinterpret_cast<float*>(Out)[(size_t)row * N + col] = v + bias[col];
          else
            reinterpret_cast<unsigned short*>(Out)[(size_t)row * N + col] = f2bf(v);
        }
      }
    }
  }
}

// ---------------------------------------------------------------------------
// Fused attention: per block = (b, h, 128-query chunk); per wave = 16 queries.
// scores 16x96 (K=160, 5 WMMA steps) -> mask+softmax in regs -> probs to LDS
// -> P@V 16x160 (K=96, 3 WMMA steps x 10 N-frags) -> bf16 out.
// ---------------------------------------------------------------------------
__global__ __launch_bounds__(256)
void attn_wmma(const unsigned short* __restrict__ qbf,  // [B*S, C]
               const unsigned short* __restrict__ kbf,  // [B*77, C]
               const unsigned short* __restrict__ vT,   // [B*H*160, 96]
               unsigned short* __restrict__ obf) {      // [B*S, C]
  __shared__ alignas(16) unsigned short Pl[8][16 * SKVP];

  const int mchunks = SS / 128;
  const int mc = blockIdx.x % mchunks;
  int t = blockIdx.x / mchunks;
  const int h = t % HH;
  const int b = t / HH;

  const int lane = threadIdx.x & 31;
  const int wave = threadIdx.x >> 5;
  const int half = lane >> 4;
  const int l16  = lane & 15;
  const int q0   = mc * 128 + wave * 16;

  const unsigned short* qp = qbf + ((size_t)(b * SS + q0)) * CC + h * DH;
  const unsigned short* kp = kbf + ((size_t)b * SENC) * CC + h * DH;
  const unsigned short* vp = vT + ((size_t)(b * HH + h)) * DH * SKVP;

  // ---- scores = q @ k^T ----
  f32x8 sacc[6];
  #pragma unroll
  for (int i = 0; i < 6; ++i) sacc[i] = fzero8();

  #pragma unroll
  for (int ks = 0; ks < 5; ++ks) {
    bf16x16 aq = ld_frag(qp + (size_t)l16 * CC + ks * 32 + half * 8);
    #pragma unroll
    for (int nf = 0; nf < 6; ++nf) {
      int n = nf * 16 + l16; if (n > SENC - 1) n = SENC - 1;  // clamp; masked later
      bf16x16 bk = ld_frag(kp + (size_t)n * CC + ks * 32 + half * 8);
      sacc[nf] = WMMA_BF16(aq, bk, sacc[nf]);
    }
  }

  // ---- mask + softmax (row = j + 8*half lives in one 16-lane half) ----
  const float scale = 0.07905694150420949f;  // 1/sqrt(160)
  unsigned short* pl = &Pl[wave][0];
  #pragma unroll
  for (int j = 0; j < 8; ++j) {
    float sv[6];
    float mx = -__builtin_inff();
    #pragma unroll
    for (int nf = 0; nf < 6; ++nf) {
      const int col = nf * 16 + l16;
      float v = sacc[nf][j] * scale;
      if (col >= SENC) v = -__builtin_inff();
      sv[nf] = v;
      mx = fmaxf(mx, v);
    }
    #pragma unroll
    for (int m = 1; m < 16; m <<= 1) mx = fmaxf(mx, __shfl_xor(mx, m, 32));
    float sum = 0.f;
    #pragma unroll
    for (int nf = 0; nf < 6; ++nf) {
      const float e = (sv[nf] == -__builtin_inff()) ? 0.f : __expf(sv[nf] - mx);
      sv[nf] = e; sum += e;
    }
    #pragma unroll
    for (int m = 1; m < 16; m <<= 1) sum += __shfl_xor(sum, m, 32);
    const float inv = 1.f / sum;
    const int row = j + half * 8;
    #pragma unroll
    for (int nf = 0; nf < 6; ++nf)
      pl[row * SKVP + nf * 16 + l16] = f2bf(sv[nf] * inv);
  }
  __syncthreads();

  // ---- out = P @ V ----
  f32x8 oacc[10];
  #pragma unroll
  for (int i = 0; i < 10; ++i) oacc[i] = fzero8();

  #pragma unroll
  for (int kc = 0; kc < 3; ++kc) {
    bf16x16 ap = ld_frag(pl + (size_t)l16 * SKVP + kc * 32 + half * 8);
    #pragma unroll
    for (int df = 0; df < 10; ++df) {
      bf16x16 bv = ld_frag(vp + (size_t)(df * 16 + l16) * SKVP + kc * 32 + half * 8);
      oacc[df] = WMMA_BF16(ap, bv, oacc[df]);
    }
  }

  #pragma unroll
  for (int df = 0; df < 10; ++df)
    #pragma unroll
    for (int j = 0; j < 8; ++j) {
      const int row = q0 + j + half * 8;
      obf[((size_t)(b * SS + row)) * CC + h * DH + df * 16 + l16] = f2bf(oacc[df][j]);
    }
}

// ---------------------------------------------------------------------------
// Host driver
// ---------------------------------------------------------------------------
extern "C" void kernel_launch(void* const* d_in, const int* in_sizes, int n_in,
                              void* d_out, int out_size, void* d_ws, size_t ws_size,
                              hipStream_t stream) {
  const float* hs     = (const float*)d_in[0];
  const float* enc    = (const float*)d_in[1];
  const float* Wq     = (const float*)d_in[2];
  const float* Wk     = (const float*)d_in[3];
  const float* Wv     = (const float*)d_in[4];
  const float* Wo     = (const float*)d_in[5];
  const float* bo     = (const float*)d_in[6];
  const float* q_down = (const float*)d_in[7];
  const float* q_up   = (const float*)d_in[8];
  const float* k_down = (const float*)d_in[9];
  const float* k_up   = (const float*)d_in[10];
  const float* v_down = (const float*)d_in[11];
  const float* v_up   = (const float*)d_in[12];
  const float* o_down = (const float*)d_in[13];
  const float* o_up   = (const float*)d_in[14];

  char* ws = (char*)d_ws;
  size_t off = 0;
  auto alloc = [&](size_t bytes) {
    void* p = ws + off;
    off = (off + bytes + 255) & ~(size_t)255;
    return p;
  };

  const size_t nHS  = (size_t)BB * SS * CC;          // 20,971,520
  const size_t nENC = (size_t)BB * SENC * CENC;      // 1,261,568
  unsigned short* Xbf  = (unsigned short*)alloc(nHS * 2);
  unsigned short* Ebf  = (unsigned short*)alloc(nENC * 2);
  unsigned short* Wqe  = (unsigned short*)alloc((size_t)CC * CC * 2);
  unsigned short* Wke  = (unsigned short*)alloc((size_t)CC * CENC * 2);
  unsigned short* Wve  = (unsigned short*)alloc((size_t)CC * CENC * 2);
  unsigned short* Woe  = (unsigned short*)alloc((size_t)CC * CC * 2);
  unsigned short* qbf  = (unsigned short*)alloc(nHS * 2);
  unsigned short* kbf  = (unsigned short*)alloc((size_t)BB * SENC * CC * 2);
  unsigned short* vbuf = (unsigned short*)alloc((size_t)BB * SENC * CC * 2);
  unsigned short* vTb  = (unsigned short*)alloc((size_t)BB * HH * DH * SKVP * 2);
  unsigned short* obf  = (unsigned short*)alloc(nHS * 2);
  (void)ws_size; (void)n_in; (void)in_sizes; (void)out_size;

  // 1) fold LoRA into weights, convert to bf16
  {
    int n = CC * CC;
    fold_weight<<<(n + 255) / 256, 256, 0, stream>>>(Wq, q_up, q_down, Wqe, CC, CC);
    fold_weight<<<(n + 255) / 256, 256, 0, stream>>>(Wo, o_up, o_down, Woe, CC, CC);
    int m = CC * CENC;
    fold_weight<<<(m + 255) / 256, 256, 0, stream>>>(Wk, k_up, k_down, Wke, CC, CENC);
    fold_weight<<<(m + 255) / 256, 256, 0, stream>>>(Wv, v_up, v_down, Wve, CC, CENC);
  }

  // 2) activations -> bf16
  f32_to_bf16<<<(int)((nHS + 255) / 256), 256, 0, stream>>>(hs, Xbf, (int)nHS);
  f32_to_bf16<<<(int)((nENC + 255) / 256), 256, 0, stream>>>(enc, Ebf, (int)nENC);

  // 3) projections (LoRA already folded)
  {
    const int Mq = BB * SS;                 // 16384
    dim3 gq(Mq / BM, CC / BN);              // 128 x 5
    gemm_bf16<<<gq, 256, 0, stream>>>(Xbf, Wqe, qbf, nullptr, Mq, CC, CC, 0);

    const int Mkv = BB * SENC;              // 1232
    dim3 gkv((Mkv + BM - 1) / BM, CC / BN); // 10 x 5
    gemm_bf16<<<gkv, 256, 0, stream>>>(Ebf, Wke, kbf, nullptr, Mkv, CC, CENC, 0);
    gemm_bf16<<<gkv, 256, 0, stream>>>(Ebf, Wve, vbuf, nullptr, Mkv, CC, CENC, 0);
  }

  // 4) per-head V transpose with zero padding to 96 keys
  {
    int n = BB * HH * DH * SKVP;
    build_vT<<<(n + 255) / 256, 256, 0, stream>>>(vbuf, vTb);
  }

  // 5) fused attention
  attn_wmma<<<BB * HH * (SS / 128), 256, 0, stream>>>(qbf, kbf, vTb, obf);

  // 6) output projection (+bias), fp32 out
  {
    const int Mo = BB * SS;
    dim3 go(Mo / BM, CC / BN);
    gemm_bf16<<<go, 256, 0, stream>>>(obf, Woe, d_out, bo, Mo, CC, CC, 1);
  }
}
